// FlashCausalSelfAttention_86431921865014
// MI455X (gfx1250) — compile-verified, matrix operand order checked by reading, not run on hardware
//
#include <hip/hip_runtime.h>

// Problem constants: B=2, S=2048, H=1024, NH=16, HD=64
#define C_B   2
#define C_S   2048
#define C_H   1024
#define C_NH  16
#define C_HD  64
#define C_BS  4096      // B*S
#define C_H3  3072      // 3*H

typedef __attribute__((ext_vector_type(16))) __bf16 v16bf;
typedef __attribute__((ext_vector_type(8)))  float  v8f;

union ABPack {
  v16bf v;
  uint4 q[2];
  unsigned short u[16];
};

__device__ __forceinline__ unsigned short f2bf(float f) {
  unsigned int x = __float_as_uint(f);
  x += 0x7FFFu + ((x >> 16) & 1u);          // round-to-nearest-even
  return (unsigned short)(x >> 16);
}

__device__ __forceinline__ void wait_ds0() {
#if __has_builtin(__builtin_amdgcn_s_wait_dscnt)
  __builtin_amdgcn_s_wait_dscnt(0);
#else
  asm volatile("s_wait_dscnt 0" ::: "memory");
#endif
}

// ---- workspace layout (units: ushort / bf16 elements) ----
static constexpr size_t OFF_XB = 0;            // [4096,1024] bf16 x
static constexpr size_t OFF_WQ = 4194304;      // [3072,1024] bf16 W_qkv^T
static constexpr size_t OFF_WO = 7340032;      // [1024,1024] bf16 W_out^T
static constexpr size_t OFF_Q  = 8388608;      // [B,NH,S,HD] bf16
static constexpr size_t OFF_K  = 12582912;     // [B,NH,S,HD] bf16
static constexpr size_t OFF_V  = 16777216;     // [B,NH,HD,S] bf16  (TRANSPOSED)
static constexpr size_t OFF_Y  = 20971520;     // [4096,1024] bf16 attention out
// total: 25165824 ushorts = 48 MB

// ---------------- Stage 0: convert / transpose to bf16 ----------------
__global__ __launch_bounds__(256) void cvt_kernel(const float* __restrict__ x,
                                                  const float* __restrict__ wqkv,
                                                  const float* __restrict__ wout,
                                                  unsigned short* __restrict__ ws) {
  size_t tid = (size_t)blockIdx.x * 256 + threadIdx.x;
  const size_t N_X = 4194304, N_WQ = 3145728;
  if (tid < N_X) {
    ws[OFF_XB + tid] = f2bf(x[tid]);
  } else if (tid < N_X + N_WQ) {
    size_t o = tid - N_X;
    int row = (int)(o >> 10);          // output column of W_qkv (0..3071)
    int k   = (int)(o & 1023);
    ws[OFF_WQ + o] = f2bf(wqkv[(size_t)k * C_H3 + row]);
  } else {
    size_t o = tid - N_X - N_WQ;
    int row = (int)(o >> 10);          // output column of W_out (0..1023)
    int k   = (int)(o & 1023);
    ws[OFF_WO + o] = f2bf(wout[(size_t)k * C_H + row]);
  }
}

// ---------------- Stage 1: QKV projection GEMM ----------------
// C[4096,3072] = Xb * WqkvT^T ; each wave owns a 32x64 tile (2x4 WMMA frags).
__global__ __launch_bounds__(256) void qkv_gemm(unsigned short* __restrict__ ws) {
  int lane = threadIdx.x & 31;
  int half = lane >> 4, lm = lane & 15;
  int wid  = blockIdx.x * 8 + (threadIdx.x >> 5);
  int mt = wid / 48;                    // 128 row tiles of 32
  int ng = wid % 48;                    // 48 col groups of 64

  const unsigned short* A0 = ws + OFF_XB + (size_t)(mt * 32 + lm) * C_H;
  const unsigned short* A1 = A0 + (size_t)16 * C_H;
  const unsigned short* Bg = ws + OFF_WQ + (size_t)(ng * 64 + lm) * C_H;

  v8f acc[2][4] = {};
  for (int k0 = 0; k0 < C_H; k0 += 32) {
    ABPack a0, a1;
    a0.q[0] = *(const uint4*)(A0 + k0 + half * 8);
    a0.q[1] = *(const uint4*)(A0 + k0 + 16 + half * 8);
    a1.q[0] = *(const uint4*)(A1 + k0 + half * 8);
    a1.q[1] = *(const uint4*)(A1 + k0 + 16 + half * 8);
#pragma unroll
    for (int nb = 0; nb < 4; ++nb) {
      ABPack b;
      const unsigned short* Bp = Bg + (size_t)(nb * 16) * C_H;
      b.q[0] = *(const uint4*)(Bp + k0 + half * 8);
      b.q[1] = *(const uint4*)(Bp + k0 + 16 + half * 8);
      acc[0][nb] = __builtin_amdgcn_wmma_f32_16x16x32_bf16(false, a0.v, false, b.v,
                                                           (short)0, acc[0][nb], false, false);
      acc[1][nb] = __builtin_amdgcn_wmma_f32_16x16x32_bf16(false, a1.v, false, b.v,
                                                           (short)0, acc[1][nb], false, false);
    }
  }

  // scatter into Q/K [B,NH,S,HD] and V [B,NH,HD,S] as bf16
  int cbase = ng * 64;
  int sec   = cbase >> 10;              // 0:Q 1:K 2:V (uniform per wave)
  int head  = (cbase & 1023) >> 6;      // uniform per wave (64-col group = 1 head)
#pragma unroll
  for (int mr = 0; mr < 2; ++mr) {
#pragma unroll
    for (int nb = 0; nb < 4; ++nb) {
      int hd = nb * 16 + lm;
#pragma unroll
      for (int e = 0; e < 8; ++e) {
        int r = mt * 32 + mr * 16 + e + half * 8;
        int bb = r >> 11, sidx = r & 2047;
        unsigned short val = f2bf(acc[mr][nb][e]);
        if (sec == 2) {
          size_t idx = (((size_t)(bb * C_NH + head) * C_HD) + hd) * C_S + sidx;
          ws[OFF_V + idx] = val;
        } else {
          size_t idx = (((size_t)(bb * C_NH + head) * C_S) + sidx) * C_HD + hd;
          ws[(sec == 0 ? OFF_Q : OFF_K) + idx] = val;
        }
      }
    }
  }
}

// ---------------- Stage 2: flash attention ----------------
// One wave per (b,h, 16-query tile); key tiles of 32; online softmax.
__global__ __launch_bounds__(256) void attn_kernel(unsigned short* __restrict__ ws) {
  __shared__ unsigned short pshare[8][16 * 32];   // per-wave P transpose buffer
  int lane  = threadIdx.x & 31;
  int half  = lane >> 4, lm = lane & 15;
  int wslot = threadIdx.x >> 5;
  int wid   = blockIdx.x * 8 + wslot;             // 4096 waves total
  int qt = wid & 127;                             // 128 query tiles
  int bh = wid >> 7;                              // 32 (b,h) pairs

  const unsigned short* Qb = ws + OFF_Q + (size_t)bh * C_S * C_HD;
  const unsigned short* Kb = ws + OFF_K + (size_t)bh * C_S * C_HD;
  const unsigned short* Vb = ws + OFF_V + (size_t)bh * C_HD * C_S;  // [HD,S]

  // Q fragments for hd-chunks 0..31 and 32..63, kept resident
  ABPack qa0, qa1;
  {
    const unsigned short* Qrow = Qb + (size_t)(qt * 16 + lm) * C_HD;
    qa0.q[0] = *(const uint4*)(Qrow + half * 8);
    qa0.q[1] = *(const uint4*)(Qrow + 16 + half * 8);
    qa1.q[0] = *(const uint4*)(Qrow + 32 + half * 8);
    qa1.q[1] = *(const uint4*)(Qrow + 48 + half * 8);
  }

  v8f oacc[4] = {v8f{}, v8f{}, v8f{}, v8f{}};
  float rmax[8], rl[8];
#pragma unroll
  for (int e = 0; e < 8; ++e) { rmax[e] = -1e30f; rl[e] = 0.f; }

  int ntiles = (qt * 16 + 15) / 32 + 1;           // causal key-tile bound
  for (int kt = 0; kt < ntiles; ++kt) {
    // ---- scores S = Q K^T (16 x 32) as two 16x16 accumulators ----
    v8f s0 = {}, s1 = {};
    {
      const unsigned short* Kr0 = Kb + (size_t)(kt * 32 + lm) * C_HD;
      const unsigned short* Kr1 = Kb + (size_t)(kt * 32 + 16 + lm) * C_HD;
      ABPack kb;
      kb.q[0] = *(const uint4*)(Kr0 + half * 8);
      kb.q[1] = *(const uint4*)(Kr0 + 16 + half * 8);
      s0 = __builtin_amdgcn_wmma_f32_16x16x32_bf16(false, qa0.v, false, kb.v, (short)0, s0, false, false);
      kb.q[0] = *(const uint4*)(Kr0 + 32 + half * 8);
      kb.q[1] = *(const uint4*)(Kr0 + 48 + half * 8);
      s0 = __builtin_amdgcn_wmma_f32_16x16x32_bf16(false, qa1.v, false, kb.v, (short)0, s0, false, false);
      kb.q[0] = *(const uint4*)(Kr1 + half * 8);
      kb.q[1] = *(const uint4*)(Kr1 + 16 + half * 8);
      s1 = __builtin_amdgcn_wmma_f32_16x16x32_bf16(false, qa0.v, false, kb.v, (short)0, s1, false, false);
      kb.q[0] = *(const uint4*)(Kr1 + 32 + half * 8);
      kb.q[1] = *(const uint4*)(Kr1 + 48 + half * 8);
      s1 = __builtin_amdgcn_wmma_f32_16x16x32_bf16(false, qa1.v, false, kb.v, (short)0, s1, false, false);
    }

    // ---- online softmax; C-layout: element e -> row, lane%16 -> key ----
    int k0g = kt * 32 + lm;
    int k1g = k0g + 16;
#pragma unroll
    for (int e = 0; e < 8; ++e) {
      int qg = qt * 16 + e + half * 8;
      float v0 = s0[e] * 0.125f; if (k0g > qg) v0 = -1e30f;   // 1/sqrt(64), causal
      float v1 = s1[e] * 0.125f; if (k1g > qg) v1 = -1e30f;
      float tmax = fmaxf(v0, v1);
      tmax = fmaxf(tmax, __shfl_xor(tmax, 1, 32));
      tmax = fmaxf(tmax, __shfl_xor(tmax, 2, 32));
      tmax = fmaxf(tmax, __shfl_xor(tmax, 4, 32));
      tmax = fmaxf(tmax, __shfl_xor(tmax, 8, 32));
      float nm = fmaxf(rmax[e], tmax);
      float alpha = __expf(rmax[e] - nm);
      float p0 = __expf(v0 - nm);
      float p1 = __expf(v1 - nm);
      float rs = p0 + p1;
      rs += __shfl_xor(rs, 1, 32);
      rs += __shfl_xor(rs, 2, 32);
      rs += __shfl_xor(rs, 4, 32);
      rs += __shfl_xor(rs, 8, 32);
      rl[e] = rl[e] * alpha + rs;
      rmax[e] = nm;
#pragma unroll
      for (int hb = 0; hb < 4; ++hb) oacc[hb][e] *= alpha;
      int row = e + half * 8;
      pshare[wslot][row * 32 + lm]      = f2bf(p0);
      pshare[wslot][row * 32 + 16 + lm] = f2bf(p1);
    }

    // ---- transpose P through wave-private LDS into A-layout ----
    wait_ds0();
    ABPack pa;
    pa.q[0] = *(const uint4*)(&pshare[wslot][lm * 32 + half * 8]);
    pa.q[1] = *(const uint4*)(&pshare[wslot][lm * 32 + 16 + half * 8]);

    // ---- O += P V : V^T layout [HD,S] -> contiguous key-runs per lane ----
#pragma unroll
    for (int hb = 0; hb < 4; ++hb) {
      ABPack vbp;
      const unsigned short* Vcol = Vb + (size_t)(hb * 16 + lm) * C_S + kt * 32;
      vbp.q[0] = *(const uint4*)(Vcol + half * 8);
      vbp.q[1] = *(const uint4*)(Vcol + 16 + half * 8);
      oacc[hb] = __builtin_amdgcn_wmma_f32_16x16x32_bf16(false, pa.v, false, vbp.v,
                                                         (short)0, oacc[hb], false, false);
    }
  }

  // ---- normalize and store Y (bf16, [B*S, H]) ----
  unsigned short* Yb = ws + OFF_Y;
  int b = bh >> 4, h = bh & 15;
#pragma unroll
  for (int hb = 0; hb < 4; ++hb) {
#pragma unroll
    for (int e = 0; e < 8; ++e) {
      float val = oacc[hb][e] / rl[e];
      int row = b * C_S + qt * 16 + e + half * 8;
      int col = h * C_HD + hb * 16 + lm;
      Yb[(size_t)row * C_H + col] = f2bf(val);
    }
  }
}

// ---------------- Stage 3: output projection ----------------
// out[4096,1024] = Y * WoutT^T ; each wave owns a 32x64 tile (2x4 WMMA frags).
__global__ __launch_bounds__(256) void out_gemm(const unsigned short* __restrict__ ws,
                                                float* __restrict__ out) {
  int lane = threadIdx.x & 31;
  int half = lane >> 4, lm = lane & 15;
  int wid  = blockIdx.x * 8 + (threadIdx.x >> 5);
  int mt = wid >> 4;                     // 128 row tiles of 32
  int ng = wid & 15;                     // 16 col groups of 64

  const unsigned short* A0 = ws + OFF_Y + (size_t)(mt * 32 + lm) * C_H;
  const unsigned short* A1 = A0 + (size_t)16 * C_H;
  const unsigned short* Bg = ws + OFF_WO + (size_t)(ng * 64 + lm) * C_H;

  v8f acc[2][4] = {};
  for (int k0 = 0; k0 < C_H; k0 += 32) {
    ABPack a0, a1;
    a0.q[0] = *(const uint4*)(A0 + k0 + half * 8);
    a0.q[1] = *(const uint4*)(A0 + k0 + 16 + half * 8);
    a1.q[0] = *(const uint4*)(A1 + k0 + half * 8);
    a1.q[1] = *(const uint4*)(A1 + k0 + 16 + half * 8);
#pragma unroll
    for (int nb = 0; nb < 4; ++nb) {
      ABPack b;
      const unsigned short* Bp = Bg + (size_t)(nb * 16) * C_H;
      b.q[0] = *(const uint4*)(Bp + k0 + half * 8);
      b.q[1] = *(const uint4*)(Bp + k0 + 16 + half * 8);
      acc[0][nb] = __builtin_amdgcn_wmma_f32_16x16x32_bf16(false, a0.v, false, b.v,
                                                           (short)0, acc[0][nb], false, false);
      acc[1][nb] = __builtin_amdgcn_wmma_f32_16x16x32_bf16(false, a1.v, false, b.v,
                                                           (short)0, acc[1][nb], false, false);
    }
  }
#pragma unroll
  for (int mr = 0; mr < 2; ++mr) {
#pragma unroll
    for (int nb = 0; nb < 4; ++nb) {
#pragma unroll
      for (int e = 0; e < 8; ++e) {
        int r = mt * 32 + mr * 16 + e + half * 8;
        int c = ng * 64 + nb * 16 + lm;
        out[(size_t)r * C_H + c] = acc[mr][nb][e];
      }
    }
  }
}

extern "C" void kernel_launch(void* const* d_in, const int* in_sizes, int n_in,
                              void* d_out, int out_size, void* d_ws, size_t ws_size,
                              hipStream_t stream) {
  (void)in_sizes; (void)n_in; (void)out_size; (void)ws_size;
  const float* x    = (const float*)d_in[0];
  const float* mask = (const float*)d_in[1];   // causal mask applied analytically
  (void)mask;
  const float* wqkv = (const float*)d_in[2];
  const float* wout = (const float*)d_in[3];
  unsigned short* ws = (unsigned short*)d_ws;
  float* out = (float*)d_out;

  cvt_kernel <<<32768, 256, 0, stream>>>(x, wqkv, wout, ws);  // 8.4M elems
  qkv_gemm   <<<768,   256, 0, stream>>>(ws);                 // 6144 wave-tiles (32x64)
  attn_kernel<<<512,   256, 0, stream>>>(ws);                 // 4096 waves
  out_gemm   <<<256,   256, 0, stream>>>(ws, out);            // 2048 wave-tiles (32x64)
}
